// MemristorConv2d_50826642981443
// MI455X (gfx1250) — matrix operand order repaired and checked by reading, not compiled, not run on hardware
//
#include <hip/hip_runtime.h>

typedef _Float16 v8h  __attribute__((ext_vector_type(8)));
typedef _Float16 v16h __attribute__((ext_vector_type(16)));
typedef float    v4f  __attribute__((ext_vector_type(4)));
typedef float    v8f  __attribute__((ext_vector_type(8)));
typedef int      v4i  __attribute__((ext_vector_type(4)));

#define ADC_SCALE_F 8020.0f
#define ADC_QMAX_F  2047.0f
#define PLANE_STRIDE (9 * 64 * 64)   // wT plane stride in f16 elements

// CDNA5 async global->LDS path, compile-time gated so this file always builds.
#if defined(__HIP_DEVICE_COMPILE__) && \
    __has_builtin(__builtin_amdgcn_global_load_async_to_lds_b128) && \
    __has_builtin(__builtin_amdgcn_s_wait_asynccnt)
#define CDNA5_ASYNC 1
#else
#define CDNA5_ASYNC 0
#endif

static __device__ __forceinline__ v16h cat16(v8h lo, v8h hi) {
  return __builtin_shufflevector(lo, hi, 0,1,2,3,4,5,6,7,8,9,10,11,12,13,14,15);
}

// ---------------------------------------------------------------------------
// Prep 1: DAC quantize + cubic nonlinearity, transpose to [b][x=w][y=h][c] f16
// in:  inputs [B=16][C=64][H=64][W=64] f32
// out: act    [16][64][64][64] f16  (channels last)
// ---------------------------------------------------------------------------
__global__ __launch_bounds__(256) void prep_act_kernel(
    const float* __restrict__ in, _Float16* __restrict__ act, int n) {
  int t = blockIdx.x * 256 + threadIdx.x;
  if (t >= n) return;
  int w = t & 63, h = (t >> 6) & 63, c = (t >> 12) & 63, b = t >> 18;
  float x = in[t];
  x = fminf(fmaxf(x, -1.0f), 1.0f);
  float vq = rintf(x * 127.0f) * (0.6f / 127.0f);   // DAC: round-to-even, VMAX=0.6
  float f  = vq + 0.1f * vq * vq * vq;              // memristor I-V nonlinearity
  act[(size_t)(((b * 64 + w) * 64 + h) << 6) + c] = (_Float16)f;
}

// ---------------------------------------------------------------------------
// Prep 2: effective conductances g_pos-g_neg -> f16, layout [m][tap][oc][c]
// so each lane's WMMA B-fragment is 16 contiguous halves (K = channel).
// ---------------------------------------------------------------------------
__global__ __launch_bounds__(256) void prep_w_kernel(
    const float* __restrict__ gp, const float* __restrict__ gn,
    _Float16* __restrict__ wt, int n) {
  int t = blockIdx.x * 256 + threadIdx.x;
  if (t >= n) return;
  int c   = t & 63;
  int oc  = (t >> 6) & 63;
  int mt  = t >> 12;          // m*9 + tap
  int tap = mt % 9;
  int m   = mt / 9;
  int i = tap / 3, j = tap % 3;
  int src = (((m * 64 + oc) * 64 + c) * 3 + i) * 3 + j;
  wt[t] = (_Float16)(gp[src] - gn[src]);
}

// ---------------------------------------------------------------------------
// Main: fused implicit-GEMM conv for all 3 bit planes + ADC + recombine + bias
// Block: 256 threads = 8 wave32 waves; tile = 32 y-positions x 64 oc, fixed (b,x)
// Wave (msub,nsub): 16x16 D tile, three f32 accumulators (one per plane).
// K = 9 taps * 64 channels, consumed in 18 chunks of 32 via v_wmma_f32_16x16x32_f16
// ---------------------------------------------------------------------------
__global__ __launch_bounds__(256) void conv_wmma_kernel(
    const _Float16* __restrict__ act, const _Float16* __restrict__ wt,
    const float* __restrict__ bias, float* __restrict__ out) {
  __shared__ _Float16 smem[3 * 34 * 64];   // act halo tile [dx=3][yy=34][c=64], 13KB

  const int tid  = threadIdx.x;
  const int lane = tid & 31;
  const int wv   = tid >> 5;      // 0..7
  const int msub = wv >> 2;       // 0..1  (y sub-tile)
  const int nsub = wv & 3;        // 0..3  (oc sub-tile)
  const int x    = blockIdx.x;    // 0..63
  const int y0b  = blockIdx.y * 32;
  const int b    = blockIdx.z;
  const int l16  = lane & 15;
  const bool hi  = lane >= 16;

  // ---- cooperative stage of padded act tile into LDS (zeros in halo) ----
  for (int cidx = tid; cidx < 816; cidx += 256) {   // 816 chunks of 8 halves
    int row  = cidx >> 3;         // 0..101 = dx*34 + yy
    int part = cidx & 7;
    int dx = row / 34;
    int yy = row % 34;
    int xa = x - 1 + dx;
    int ya = y0b - 1 + yy;
    _Float16* sdst = smem + (row << 6) + part * 8;
    if (xa >= 0 && xa < 64 && ya >= 0 && ya < 64) {
      const _Float16* gsrc =
          act + (size_t)(((b * 64 + xa) * 64 + ya) << 6) + part * 8;
#if CDNA5_ASYNC
      __builtin_amdgcn_global_load_async_to_lds_b128(
          (__attribute__((address_space(1))) v4i*)(void*)gsrc,
          (__attribute__((address_space(3))) v4i*)(void*)sdst, 0, 0);
#else
      *(v8h*)sdst = *(const v8h*)gsrc;
#endif
    } else {
      *(v8h*)sdst = (v8h){};      // zero halo (padding='same')
    }
  }
#if CDNA5_ASYNC
  __builtin_amdgcn_s_wait_asynccnt(0);  // async-LDS writes ordered before barrier
#endif
  __syncthreads();

  v8f acc0 = {}, acc1 = {}, acc2 = {};

  // 16-bit A 16x32 layout (matches B): lanes 0-15 take K 0..15, lanes 16-31 take K 16..31
  const int aoff = hi ? 16 : 0;
  // 16-bit B 32x16 layout (matches A): lanes 0-15 take K 0..15, lanes 16-31 take K 16..31
  const int boff = hi ? 16 : 0;
  const int oc   = nsub * 16 + l16;
  const int arow = msub * 16 + l16;   // in-tile y (tap j adds 0..2 onto halo base)

  for (int tap = 0; tap < 9; ++tap) {
    const int i = tap / 3, j = tap % 3;
    const _Float16* abase = smem + ((i * 34 + arow + j) << 6);
    const _Float16* wbase = wt + ((tap * 64 + oc) << 6) + boff;
    if (tap < 8)  // warm L2/WGP$ for next tap's weight block
      __builtin_prefetch(wbase + (1 << 12), 0, 1);
#pragma unroll
    for (int ch = 0; ch < 2; ++ch) {
      const int cb = ch * 32;
      v16h A  = cat16(*(const v8h*)(abase + cb + aoff),
                      *(const v8h*)(abase + cb + 8 + aoff));
      v16h B0 = cat16(*(const v8h*)(wbase + cb),
                      *(const v8h*)(wbase + cb + 8));
      v16h B1 = cat16(*(const v8h*)(wbase + PLANE_STRIDE + cb),
                      *(const v8h*)(wbase + PLANE_STRIDE + cb + 8));
      v16h B2 = cat16(*(const v8h*)(wbase + 2 * PLANE_STRIDE + cb),
                      *(const v8h*)(wbase + 2 * PLANE_STRIDE + cb + 8));
      acc0 = __builtin_amdgcn_wmma_f32_16x16x32_f16(false, A, false, B0, (short)0, acc0, false, false);
      acc1 = __builtin_amdgcn_wmma_f32_16x16x32_f16(false, A, false, B1, (short)0, acc1, false, false);
      acc2 = __builtin_amdgcn_wmma_f32_16x16x32_f16(false, A, false, B2, (short)0, acc2, false, false);
    }
  }

  // ---- epilogue: per-plane ADC, recombine 4*a0 + 2*a1 + 2*a2, bias, store ----
  // D layout: VGPR r -> M = r (+8 for lanes 16-31), N = lane%16
  // Each lane's 8 results are y-contiguous & 32B aligned -> two b128 stores.
  const float bs = bias[oc];
  float* obase = out + (size_t)(((b * 64 + oc) * 64 + x) << 6)
               + y0b + msub * 16 + (hi ? 8 : 0);
  float res[8];
#pragma unroll
  for (int r = 0; r < 8; ++r) {
    float q0 = fminf(fmaxf(rintf(acc0[r] * ADC_SCALE_F), -ADC_QMAX_F), ADC_QMAX_F);
    float q1 = fminf(fmaxf(rintf(acc1[r] * ADC_SCALE_F), -ADC_QMAX_F), ADC_QMAX_F);
    float q2 = fminf(fmaxf(rintf(acc2[r] * ADC_SCALE_F), -ADC_QMAX_F), ADC_QMAX_F);
    res[r] = (4.0f * q0 + 2.0f * q1 + 2.0f * q2) * (1.0f / ADC_SCALE_F) + bs;
  }
  *(v4f*)(obase + 0) = (v4f){res[0], res[1], res[2], res[3]};
  *(v4f*)(obase + 4) = (v4f){res[4], res[5], res[6], res[7]};
}

// ---------------------------------------------------------------------------
extern "C" void kernel_launch(void* const* d_in, const int* in_sizes, int n_in,
                              void* d_out, int out_size, void* d_ws, size_t ws_size,
                              hipStream_t stream) {
  const float* inputs = (const float*)d_in[0];  // [16,64,64,64]
  const float* g_pos  = (const float*)d_in[1];  // [3,64,64,3,3]
  const float* g_neg  = (const float*)d_in[2];  // [3,64,64,3,3]
  const float* bias   = (const float*)d_in[3];  // [64]
  float* out = (float*)d_out;                   // [16,64,64,64] f32

  const size_t act_elems = (size_t)16 * 64 * 64 * 64;       // 4M f16 = 8 MB
  _Float16* act = (_Float16*)d_ws;
  _Float16* wt  = (_Float16*)((char*)d_ws + act_elems * sizeof(_Float16));

  int nact = (int)act_elems;
  prep_act_kernel<<<(nact + 255) / 256, 256, 0, stream>>>(inputs, act, nact);

  int nw = 3 * 9 * 64 * 64;                                  // 110592
  prep_w_kernel<<<(nw + 255) / 256, 256, 0, stream>>>(g_pos, g_neg, wt, nw);

  dim3 grid(64, 2, 16);   // x, y-half, batch
  conv_wmma_kernel<<<grid, 256, 0, stream>>>(act, wt, bias, out);
}
